// EdgeConv_63032940036480
// MI455X (gfx1250) — matrix operand order repaired and checked by reading, not compiled, not run on hardware
//
#include <hip/hip_runtime.h>

#define B_ 16
#define C_ 64
#define N_ 2048
#define O_ 64
#define K_ 20

typedef __bf16 bf16;
typedef bf16 v16bf __attribute__((ext_vector_type(16)));
typedef float v8f   __attribute__((ext_vector_type(8)));

__device__ __forceinline__ void split_bf16(float v, bf16& hi, bf16& lo) {
    hi = (bf16)v;
    lo = (bf16)(v - (float)hi);
}

// ---------------------------------------------------------------------------
// xx[b,n] = sum_c x[b,c,n]^2
// ---------------------------------------------------------------------------
__global__ __launch_bounds__(256) void ec_xx_kernel(const float* __restrict__ x,
                                                    float* __restrict__ xx) {
    int i = blockIdx.x * 256 + threadIdx.x;          // over B*N
    int b = i >> 11, n = i & (N_ - 1);
    float s = 0.f;
    #pragma unroll 8
    for (int c = 0; c < C_; ++c) {
        float v = x[(b * C_ + c) * N_ + n];
        s += v * v;
    }
    xx[i] = s;
}

// ---------------------------------------------------------------------------
// Projections: z1[b,n,o] = W1[o,:]·x[b,:,n], z2[b,n,o] = (W2-W1)[o,:]·x[b,:,n]
// Stacked A = [W1 ; W2-W1] (128 x 64), bf16-split WMMA (hi*hi + hi*lo + lo*hi)
// ---------------------------------------------------------------------------
__global__ __launch_bounds__(128) void ec_proj_kernel(const float* __restrict__ x,
                                                      const float* __restrict__ W,
                                                      float* __restrict__ z1,
                                                      float* __restrict__ z2) {
    __shared__ bf16 Whi[128][66], Wlo[128][66];
    __shared__ bf16 Xhi[64][66],  Xlo[64][66];
    const int t  = threadIdx.x;
    const int b  = blockIdx.y;
    const int nb = blockIdx.x * 64;

    for (int i = t; i < 128 * 64; i += 128) {
        int mo = i & 127, c = i >> 7;
        int o  = mo & 63;
        float w0 = W[o * 128 + c];
        float w1 = W[o * 128 + 64 + c];
        float v  = (mo < 64) ? w0 : (w1 - w0);
        bf16 hi, lo; split_bf16(v, hi, lo);
        Whi[mo][c] = hi; Wlo[mo][c] = lo;
    }
    for (int i = t; i < 64 * 64; i += 128) {
        int n = i & 63, c = i >> 6;
        float v = x[(b * C_ + c) * N_ + nb + n];
        bf16 hi, lo; split_bf16(v, hi, lo);
        Xhi[c][n] = hi; Xlo[c][n] = lo;
    }
    __syncthreads();

    const int w = t >> 5, lane = t & 31, l = lane & 15, h = lane >> 4;
    for (int s = w; s < 32; s += 4) {               // 8 M-subtiles x 4 N-subtiles
        const int RM = (s >> 2) * 16, CN = (s & 3) * 16;
        v8f acc = {};
        #pragma unroll
        for (int kb = 0; kb < 64; kb += 32) {
            v16bf ah, al, bh, bl;
            #pragma unroll
            for (int e = 0; e < 16; ++e) {           // A: 16-bit 16x32 layout
                int kk = kb + ((e < 8) ? (h * 8 + e) : (16 + h * 8 + (e - 8)));
                ah[e] = Whi[RM + l][kk];
                al[e] = Wlo[RM + l][kk];
            }
            #pragma unroll
            for (int e = 0; e < 16; ++e) {           // B: 16-bit 32x16 layout
                int kk = kb + ((e < 8) ? (h * 8 + e) : (16 + h * 8 + (e - 8)));
                bh[e] = Xhi[kk][CN + l];
                bl[e] = Xlo[kk][CN + l];
            }
            acc = __builtin_amdgcn_wmma_f32_16x16x32_bf16(false, ah, false, bh, (short)0, acc, false, false);
            acc = __builtin_amdgcn_wmma_f32_16x16x32_bf16(false, ah, false, bl, (short)0, acc, false, false);
            acc = __builtin_amdgcn_wmma_f32_16x16x32_bf16(false, al, false, bh, (short)0, acc, false, false);
        }
        #pragma unroll
        for (int r = 0; r < 8; ++r) {
            int mo = RM + r + 8 * h;
            int n  = nb + CN + l;
            float v = acc[r];
            if (mo < 64) z1[(b * N_ + n) * O_ + mo]        = v;
            else         z2[(b * N_ + n) * O_ + (mo - 64)] = v;
        }
    }
}

// ---------------------------------------------------------------------------
// Fused distance + top-K. 64 rows per block (1 row per thread), column tiles
// of 64. neg_dist tile = 2*inner - xx_i - xx_j computed via bf16-split WMMA
// into padded LDS, then each thread streams its row into a 20-deep register
// insertion list (strict > matches JAX tie-break: lower index first).
// Next column tile is prefetched (global_prefetch_b8) while the current one
// is multiplied and scanned.
// ---------------------------------------------------------------------------
__global__ __launch_bounds__(64) void ec_knn_kernel(const float* __restrict__ x,
                                                    const float* __restrict__ xx,
                                                    int* __restrict__ idxp) {
    __shared__ bf16 Ahi[64][66], Alo[64][66];
    __shared__ bf16 Bhi[64][66], Blo[64][66];
    __shared__ float Stile[64][65];
    __shared__ float xxI[64], xxJ[64];

    const int t  = threadIdx.x;
    const int b  = blockIdx.y;
    const int rb = blockIdx.x * 64;

    for (int i = t; i < 64 * 64; i += 64) {          // stage row block (A)
        int p = i & 63, c = i >> 6;
        float v = x[(b * C_ + c) * N_ + rb + p];
        bf16 hi, lo; split_bf16(v, hi, lo);
        Ahi[p][c] = hi; Alo[p][c] = lo;
    }
    xxI[t] = xx[b * N_ + rb + t];

    float tv[K_]; int ti[K_];
    #pragma unroll
    for (int k = 0; k < K_; ++k) { tv[k] = -3.4e38f; ti[k] = 0; }

    const int w = t >> 5, lane = t & 31, l = lane & 15, h = lane >> 4;

    for (int cb = 0; cb < N_; cb += 64) {
        __syncthreads();                              // prev readers done
        for (int i = t; i < 64 * 64; i += 64) {       // stage col block (B)
            int n = i & 63, c = i >> 6;
            float v = x[(b * C_ + c) * N_ + cb + n];
            bf16 hi, lo; split_bf16(v, hi, lo);
            Bhi[c][n] = hi; Blo[c][n] = lo;
        }
        xxJ[t] = xx[b * N_ + cb + t];

        // Prefetch next 64-point column tile: thread t owns channel row c=t
        // (64 floats = 256B = 2 cachelines) -> global_prefetch_b8
        if (cb + 64 < N_) {
            const float* np = x + (b * C_ + t) * N_ + cb + 64;
            __builtin_prefetch(np, 0, 3);
            __builtin_prefetch(np + 32, 0, 3);
            if (t == 0) __builtin_prefetch(xx + b * N_ + cb + 64, 0, 3);
        }
        __syncthreads();

        for (int s = w; s < 16; s += 2) {             // 4x4 subtiles, 2 waves
            const int RM = (s >> 2) * 16, CN = (s & 3) * 16;
            v8f acc = {};
            #pragma unroll
            for (int kb = 0; kb < 64; kb += 32) {
                v16bf ah, al, bh, bl;
                #pragma unroll
                for (int e = 0; e < 16; ++e) {
                    int kk = kb + ((e < 8) ? (h * 8 + e) : (16 + h * 8 + (e - 8)));
                    ah[e] = Ahi[RM + l][kk];
                    al[e] = Alo[RM + l][kk];
                }
                #pragma unroll
                for (int e = 0; e < 16; ++e) {
                    int kk = kb + ((e < 8) ? (h * 8 + e) : (16 + h * 8 + (e - 8)));
                    bh[e] = Bhi[kk][CN + l];
                    bl[e] = Blo[kk][CN + l];
                }
                acc = __builtin_amdgcn_wmma_f32_16x16x32_bf16(false, ah, false, bh, (short)0, acc, false, false);
                acc = __builtin_amdgcn_wmma_f32_16x16x32_bf16(false, ah, false, bl, (short)0, acc, false, false);
                acc = __builtin_amdgcn_wmma_f32_16x16x32_bf16(false, al, false, bh, (short)0, acc, false, false);
            }
            #pragma unroll
            for (int r = 0; r < 8; ++r) {
                int row = RM + r + 8 * h, col = CN + l;
                Stile[row][col] = 2.0f * acc[r] - xxI[row] - xxJ[col];
            }
        }
        __syncthreads();

        #pragma unroll 4
        for (int j = 0; j < 64; ++j) {                // row scan, top-K update
            float v = Stile[t][j];
            if (v > tv[K_ - 1]) {
                float cv = v; int ci = cb + j;
                #pragma unroll
                for (int k = 0; k < K_; ++k) {
                    if (cv > tv[k]) {
                        float tf = tv[k]; int tx = ti[k];
                        tv[k] = cv; ti[k] = ci; cv = tf; ci = tx;
                    }
                }
            }
        }
    }
    int* op = idxp + (b * N_ + rb + t) * K_;
    #pragma unroll
    for (int k = 0; k < K_; ++k) op[k] = ti[k];
}

// ---------------------------------------------------------------------------
// BN statistics: per-channel sum / sumsq over all (b,n,k) of y = z1[m]+z2[n]
// ---------------------------------------------------------------------------
__global__ __launch_bounds__(256) void ec_stats_kernel(const float* __restrict__ z1,
                                                       const float* __restrict__ z2,
                                                       const int* __restrict__ idxp,
                                                       float* __restrict__ gsum,
                                                       float* __restrict__ gsum2) {
    __shared__ float r1[256], r2[256];
    const int t = threadIdx.x;
    const int o = t & 63, g = t >> 6;
    const int b = blockIdx.y;
    const int n0 = blockIdx.x * 64;
    float s = 0.f, s2 = 0.f;
    for (int nn = g; nn < 64; nn += 4) {
        int n = n0 + nn;
        float zc = z2[(b * N_ + n) * O_ + o];
        const int* ip = idxp + (b * N_ + n) * K_;
        #pragma unroll 4
        for (int k = 0; k < K_; ++k) {
            int m = ip[k];
            float y = z1[(b * N_ + m) * O_ + o] + zc;
            s += y; s2 += y * y;
        }
    }
    r1[t] = s; r2[t] = s2;
    __syncthreads();
    if (t < 64) {
        s  = r1[t] + r1[t + 64] + r1[t + 128] + r1[t + 192];
        s2 = r2[t] + r2[t + 64] + r2[t + 128] + r2[t + 192];
        atomicAdd(&gsum[t], s);
        atomicAdd(&gsum2[t], s2);
    }
}

__global__ __launch_bounds__(64) void ec_finalize_kernel(const float* __restrict__ gsum,
                                                         const float* __restrict__ gsum2,
                                                         const float* __restrict__ gamma,
                                                         const float* __restrict__ beta,
                                                         float* __restrict__ scale,
                                                         float* __restrict__ shift) {
    int o = threadIdx.x;
    const float inv_cnt = 1.0f / (float)(B_ * N_ * K_);
    float mean = gsum[o] * inv_cnt;
    float var  = gsum2[o] * inv_cnt - mean * mean;
    float sc   = gamma[o] * rsqrtf(var + 1e-5f);
    scale[o] = sc;
    shift[o] = beta[o] - mean * sc;
}

// ---------------------------------------------------------------------------
// out[b,o,n] = max_k leaky(scale[o]*(z1[m_k]+z2[n]) + shift[o])
// ---------------------------------------------------------------------------
__global__ __launch_bounds__(256) void ec_out_kernel(const float* __restrict__ z1,
                                                     const float* __restrict__ z2,
                                                     const int* __restrict__ idxp,
                                                     const float* __restrict__ scale,
                                                     const float* __restrict__ shift,
                                                     float* __restrict__ out) {
    const int n = blockIdx.x * 256 + threadIdx.x;
    const int o = blockIdx.y;
    const int b = blockIdx.z;
    const float sc = scale[o], sh = shift[o];
    const float zc = z2[(b * N_ + n) * O_ + o];
    const int* ip = idxp + (b * N_ + n) * K_;
    float mx = -3.4e38f;
    #pragma unroll 4
    for (int k = 0; k < K_; ++k) {
        int m = ip[k];
        float y = z1[(b * N_ + m) * O_ + o] + zc;
        float v = fmaf(y, sc, sh);
        v = (v >= 0.f) ? v : 0.2f * v;
        mx = fmaxf(mx, v);
    }
    out[(b * O_ + o) * N_ + n] = mx;
}

// ---------------------------------------------------------------------------
extern "C" void kernel_launch(void* const* d_in, const int* in_sizes, int n_in,
                              void* d_out, int out_size, void* d_ws, size_t ws_size,
                              hipStream_t stream) {
    const float* x     = (const float*)d_in[0];
    const float* W     = (const float*)d_in[1];
    const float* gamma = (const float*)d_in[2];
    const float* beta  = (const float*)d_in[3];
    float* out = (float*)d_out;

    float* wsf   = (float*)d_ws;
    float* z1    = wsf;                                   // 2,097,152 f
    float* z2    = wsf + 2097152;                         // 2,097,152 f
    int*   idxp  = (int*)(wsf + 4194304);                 //   655,360 i
    float* xxp   = wsf + 4849664;                         //    32,768 f
    float* gsum  = wsf + 4882432;                         //        64 f
    float* gsum2 = wsf + 4882496;                         //        64 f
    float* scale = wsf + 4882560;                         //        64 f
    float* shift = wsf + 4882624;                         //        64 f

    hipMemsetAsync(gsum, 0, 2 * O_ * sizeof(float), stream);

    ec_xx_kernel  <<<dim3((B_ * N_) / 256), 256, 0, stream>>>(x, xxp);
    ec_proj_kernel<<<dim3(N_ / 64, B_),     128, 0, stream>>>(x, W, z1, z2);
    ec_knn_kernel <<<dim3(N_ / 64, B_),      64, 0, stream>>>(x, xxp, idxp);
    ec_stats_kernel<<<dim3(N_ / 64, B_),    256, 0, stream>>>(z1, z2, idxp, gsum, gsum2);
    ec_finalize_kernel<<<1, 64, 0, stream>>>(gsum, gsum2, gamma, beta, scale, shift);
    ec_out_kernel <<<dim3(N_ / 256, O_, B_), 256, 0, stream>>>(z1, z2, idxp, scale, shift, out);
}
